// AltitudeConditionedMoE_45672682226014
// MI455X (gfx1250) — matrix operand order repaired and checked by reading, not compiled, not run on hardware
//
#include <hip/hip_runtime.h>
#include <hip/hip_bf16.h>

#define D      1024
#define EXPERTS 8
#define ALT    32
#define DFF    4096
#define BB     8
#define NN     1024
#define T      (BB*NN)      // 8192 tokens
#define LN_EPS 1e-5f

#define KSTEP  64
#define LROW   72           // padded LDS row stride in elements (144B: bank-conflict-free)

typedef __attribute__((ext_vector_type(16))) __bf16 v16bf;
typedef __attribute__((ext_vector_type(8)))  float  v8f;

union FragBF { v16bf bf; uint4 q[2]; };

__device__ __forceinline__ unsigned short f2bf(float x) {
    unsigned int u = __float_as_uint(x);
    unsigned int r = u + 0x7fffu + ((u >> 16) & 1u);   // round-to-nearest-even
    return (unsigned short)(r >> 16);
}
__device__ __forceinline__ float gelu_exact(float x) {
    return 0.5f * x * (1.0f + erff(x * 0.70710678118654752f));
}

// gfx1250 async copy global -> LDS (16 bytes/lane), tracked by ASYNCcnt
__device__ __forceinline__ void async_b128(unsigned lds_addr, unsigned long long gaddr) {
    asm volatile("global_load_async_to_lds_b128 %0, %1, off"
                 :: "v"(lds_addr), "v"(gaddr) : "memory");
}

// ---------------------------------------------------------------- small init
__global__ void init_small_kernel(int* counts, float* f_acc, float* p_acc) {
    int i = threadIdx.x;
    if (i < EXPERTS) { counts[i] = 0; f_acc[i] = 0.f; p_acc[i] = 0.f; }
}

// residual init: out = tokens
__global__ void copy_tokens_kernel(const float* __restrict__ in, float* __restrict__ out, int n) {
    int i = blockIdx.x * blockDim.x + threadIdx.x;
    for (; i < n; i += gridDim.x * blockDim.x) out[i] = in[i];
}

// ------------------------------------------- transpose f32 [R,C] -> bf16 [C,R]
__global__ void transpose_f32_to_bf16(const float* __restrict__ in,
                                      unsigned short* __restrict__ out,
                                      int R, int C) {
    __shared__ float tile[32][33];
    int e = blockIdx.z;
    const float* src = in + (size_t)e * R * C;
    unsigned short* dst = out + (size_t)e * C * R;
    int c0 = blockIdx.x * 32, r0 = blockIdx.y * 32;
    int tx = threadIdx.x, ty = threadIdx.y;            // 32 x 8
    for (int i = 0; i < 32; i += 8)
        tile[ty + i][tx] = src[(size_t)(r0 + ty + i) * C + (c0 + tx)];
    __syncthreads();
    for (int i = 0; i < 32; i += 8)
        dst[(size_t)(c0 + ty + i) * R + (r0 + tx)] = f2bf(tile[tx][ty + i]);
}

// -------------------------------------- LayerNorm + router + top-2 + routing
__global__ __launch_bounds__(256) void ln_router_kernel(
    const float* __restrict__ tokens, const float* __restrict__ alt_emb,
    const float* __restrict__ gamma,  const float* __restrict__ beta,
    const float* __restrict__ gw1,    const float* __restrict__ gb1,
    const float* __restrict__ gw2,    const float* __restrict__ gb2,
    unsigned short* __restrict__ normed_bf,
    float* __restrict__ combine, int* __restrict__ counts, int* __restrict__ lists,
    float* __restrict__ f_acc, float* __restrict__ p_acc)
{
    __shared__ float sN[D];
    __shared__ float sA[ALT];
    __shared__ float sH[D / 2];
    __shared__ float red[256];
    __shared__ float sL[EXPERTS];
    __shared__ float sStat[2];

    int t = blockIdx.x;
    int tid = threadIdx.x;
    const float* x = tokens + (size_t)t * D;
    int b = t >> 10;                                   // t / NN

    float s = 0.f, s2 = 0.f;
    for (int j = tid; j < D; j += 256) { float v = x[j]; s += v; s2 += v * v; }
    red[tid] = s; __syncthreads();
    for (int o = 128; o > 0; o >>= 1) { if (tid < o) red[tid] += red[tid + o]; __syncthreads(); }
    if (tid == 0) sStat[0] = red[0] * (1.f / D);
    __syncthreads();
    red[tid] = s2; __syncthreads();
    for (int o = 128; o > 0; o >>= 1) { if (tid < o) red[tid] += red[tid + o]; __syncthreads(); }
    if (tid == 0) { float mu = sStat[0]; sStat[1] = red[0] * (1.f / D) - mu * mu; }
    __syncthreads();
    float mu = sStat[0];
    float inv = rsqrtf(sStat[1] + LN_EPS);
    for (int j = tid; j < D; j += 256) {
        float nv = (x[j] - mu) * inv * gamma[j] + beta[j];
        sN[j] = nv;
        normed_bf[(size_t)t * D + j] = f2bf(nv);
    }
    if (tid < ALT) sA[tid] = alt_emb[b * ALT + tid];
    __syncthreads();

    for (int j = tid; j < D / 2; j += 256) {
        float acc = gb1[j];
        for (int i = 0; i < D;   i++) acc += sN[i] * gw1[i * (D / 2) + j];
        for (int i = 0; i < ALT; i++) acc += sA[i] * gw1[(D + i) * (D / 2) + j];
        sH[j] = gelu_exact(acc);
    }
    __syncthreads();
    if (tid < EXPERTS) {
        float acc = gb2[tid];
        for (int i = 0; i < D / 2; i++) acc += sH[i] * gw2[i * EXPERTS + tid];
        sL[tid] = acc;
    }
    __syncthreads();
    if (tid == 0) {
        int i0 = 0;
        for (int e = 1; e < EXPERTS; e++) if (sL[e] > sL[i0]) i0 = e;
        int i1 = -1;
        for (int e = 0; e < EXPERTS; e++) {
            if (e == i0) continue;
            if (i1 < 0 || sL[e] > sL[i1]) i1 = e;
        }
        float v0 = sL[i0], v1 = sL[i1];
        float e1 = expf(v1 - v0);                      // v0 is max
        float g0 = 1.f / (1.f + e1), g1 = e1 / (1.f + e1);
        for (int e = 0; e < EXPERTS; e++) combine[(size_t)t * EXPERTS + e] = 0.f;
        combine[(size_t)t * EXPERTS + i0] = g0;
        combine[(size_t)t * EXPERTS + i1] = g1;
        // load-balance stats
        float mm = sL[0];
        for (int e = 1; e < EXPERTS; e++) mm = fmaxf(mm, sL[e]);
        float den = 0.f, pr[EXPERTS];
        for (int e = 0; e < EXPERTS; e++) { pr[e] = expf(sL[e] - mm); den += pr[e]; }
        for (int e = 0; e < EXPERTS; e++) atomicAdd(&p_acc[e], pr[e] / den);
        atomicAdd(&f_acc[i0], 1.0f);
        // routing lists
        int s0 = atomicAdd(&counts[i0], 1); lists[i0 * T + s0] = t;
        int s1 = atomicAdd(&counts[i1], 1); lists[i1 * T + s1] = t;
    }
}

// ----------------------------------------------- expert GEMM1: h = gelu(A*W1+b1)
// A = gathered normed rows [cnt, D] bf16 ; W1^T stored [E, DFF, D] bf16
__global__ __launch_bounds__(256) void expert_gemm1_kernel(
    const unsigned short* __restrict__ normed_bf,
    const unsigned short* __restrict__ w1t,
    const float* __restrict__ eb1,
    const int* __restrict__ counts, const int* __restrict__ lists,
    unsigned short* __restrict__ h, int e)
{
    int cnt = counts[e];
    int mBase = blockIdx.y * 128;
    if (mBase >= cnt) return;
    int nBase = blockIdx.x * 128;

    __shared__ int sTok[128];
    __shared__ unsigned short ldsA[2][128 * LROW];   // [row][k], padded rows
    __shared__ unsigned short ldsB[2][128 * LROW];   // [col][k]
    int tid = threadIdx.x;
    if (tid < 128) {
        int slot = mBase + tid;
        int cs = slot < cnt ? slot : cnt - 1;
        sTok[tid] = lists[e * T + cs];
    }
    __syncthreads();

    int w = tid >> 5, lane = tid & 31;
    int wr = w >> 1, wc = w & 1;                       // 4x2 wave grid: 32M x 64N each
    int r0 = lane & 15, half = lane >> 4;
    const unsigned short* w1e = w1t + (size_t)e * DFF * D;

    // staging: thread copies 64B of A-tile row (tid>>1) and 64B of B-tile col (tid>>1)
    int cr = tid >> 1, ch = (tid & 1) * 32;            // 32 elements per half
    const unsigned short* aSrc = normed_bf + (size_t)sTok[cr] * D + ch;
    const unsigned short* bSrc = w1e + (size_t)(nBase + cr) * D + ch;
    unsigned ldsAaddr[2], ldsBaddr[2];
    for (int bfi = 0; bfi < 2; bfi++) {
        ldsAaddr[bfi] = (unsigned)(uintptr_t)&ldsA[bfi][cr * LROW + ch];
        ldsBaddr[bfi] = (unsigned)(uintptr_t)&ldsB[bfi][cr * LROW + ch];
    }
    auto stage = [&](int bfi, int k0) {
        unsigned long long ga = (unsigned long long)(uintptr_t)(aSrc + k0);
        unsigned long long gb = (unsigned long long)(uintptr_t)(bSrc + k0);
        async_b128(ldsAaddr[bfi],      ga);
        async_b128(ldsAaddr[bfi] + 16, ga + 16);
        async_b128(ldsAaddr[bfi] + 32, ga + 32);
        async_b128(ldsAaddr[bfi] + 48, ga + 48);
        async_b128(ldsBaddr[bfi],      gb);
        async_b128(ldsBaddr[bfi] + 16, gb + 16);
        async_b128(ldsBaddr[bfi] + 32, gb + 32);
        async_b128(ldsBaddr[bfi] + 48, gb + 48);
        __builtin_prefetch(bSrc + k0 + 2 * KSTEP, 0, 1);
    };

    // per-lane fragment element offsets inside a 128 x KSTEP (padded) LDS tile
    int aOff[2], bOff[4];
    for (int mi = 0; mi < 2; mi++) aOff[mi] = (wr * 32 + mi * 16 + r0) * LROW;
    for (int ni = 0; ni < 4; ni++) bOff[ni] = (wc * 64 + ni * 16 + r0) * LROW;

    v8f acc[2][4];
    for (int mi = 0; mi < 2; mi++)
        for (int ni = 0; ni < 4; ni++)
            acc[mi][ni] = (v8f){0.f, 0.f, 0.f, 0.f, 0.f, 0.f, 0.f, 0.f};

    const int NK = D / KSTEP;
    stage(0, 0);
    for (int kk = 0; kk < NK; kk++) {
        int cur = kk & 1;
        if (kk + 1 < NK) {
            stage(cur ^ 1, (kk + 1) * KSTEP);
            asm volatile("s_wait_asynccnt 0x8" ::: "memory");  // previous stage landed
        } else {
            asm volatile("s_wait_asynccnt 0x0" ::: "memory");
        }
        __syncthreads();
        for (int ks = 0; ks < 2; ks++) {                       // two 32-K sub-steps
            int kso = ks * 32;
            FragBF a[2], bfr[4];
            for (int mi = 0; mi < 2; mi++) {
                a[mi].q[0] = *(const uint4*)&ldsA[cur][aOff[mi] + kso + half * 8];
                a[mi].q[1] = *(const uint4*)&ldsA[cur][aOff[mi] + kso + 16 + half * 8];
            }
            for (int ni = 0; ni < 4; ni++) {
                bfr[ni].q[0] = *(const uint4*)&ldsB[cur][bOff[ni] + kso + half * 8];
                bfr[ni].q[1] = *(const uint4*)&ldsB[cur][bOff[ni] + kso + 16 + half * 8];
            }
            for (int ni = 0; ni < 4; ni++)
                for (int mi = 0; mi < 2; mi++)
                    acc[mi][ni] = __builtin_amdgcn_wmma_f32_16x16x32_bf16(
                        false, a[mi].bf, false, bfr[ni].bf, (short)0, acc[mi][ni], false, false);
        }
        __syncthreads();
    }

    for (int mi = 0; mi < 2; mi++)
        for (int ni = 0; ni < 4; ni++) {
            int n = nBase + wc * 64 + ni * 16 + r0;
            float bias = eb1[e * DFF + n];
            for (int i = 0; i < 8; i++) {
                int l = wr * 32 + mi * 16 + i + 8 * half;
                int slot = mBase + l;
                if (slot < cnt)
                    h[(size_t)slot * DFF + n] = f2bf(gelu_exact(acc[mi][ni][i] + bias));
            }
        }
}

// --------------------------------- expert GEMM2: out[tok] += g*(h*W2 + b2)
// h [cnt, DFF] bf16 ; W2^T stored [E, D, DFF] bf16
__global__ __launch_bounds__(256) void expert_gemm2_kernel(
    const unsigned short* __restrict__ h,
    const unsigned short* __restrict__ w2t,
    const float* __restrict__ eb2,
    const float* __restrict__ combine,
    const int* __restrict__ counts, const int* __restrict__ lists,
    float* __restrict__ out, int e)
{
    int cnt = counts[e];
    int mBase = blockIdx.y * 128;
    if (mBase >= cnt) return;
    int nBase = blockIdx.x * 128;

    __shared__ int sTok[128];
    __shared__ int sSlot[128];
    __shared__ unsigned short ldsA[2][128 * LROW];
    __shared__ unsigned short ldsB[2][128 * LROW];
    int tid = threadIdx.x;
    if (tid < 128) {
        int slot = mBase + tid;
        int cs = slot < cnt ? slot : cnt - 1;
        sSlot[tid] = cs;
        sTok[tid] = lists[e * T + cs];
    }
    __syncthreads();

    int w = tid >> 5, lane = tid & 31;
    int wr = w >> 1, wc = w & 1;
    int r0 = lane & 15, half = lane >> 4;
    const unsigned short* w2e = w2t + (size_t)e * D * DFF;

    int cr = tid >> 1, ch = (tid & 1) * 32;
    const unsigned short* aSrc = h + (size_t)sSlot[cr] * DFF + ch;
    const unsigned short* bSrc = w2e + (size_t)(nBase + cr) * DFF + ch;
    unsigned ldsAaddr[2], ldsBaddr[2];
    for (int bfi = 0; bfi < 2; bfi++) {
        ldsAaddr[bfi] = (unsigned)(uintptr_t)&ldsA[bfi][cr * LROW + ch];
        ldsBaddr[bfi] = (unsigned)(uintptr_t)&ldsB[bfi][cr * LROW + ch];
    }
    auto stage = [&](int bfi, int k0) {
        unsigned long long ga = (unsigned long long)(uintptr_t)(aSrc + k0);
        unsigned long long gb = (unsigned long long)(uintptr_t)(bSrc + k0);
        async_b128(ldsAaddr[bfi],      ga);
        async_b128(ldsAaddr[bfi] + 16, ga + 16);
        async_b128(ldsAaddr[bfi] + 32, ga + 32);
        async_b128(ldsAaddr[bfi] + 48, ga + 48);
        async_b128(ldsBaddr[bfi],      gb);
        async_b128(ldsBaddr[bfi] + 16, gb + 16);
        async_b128(ldsBaddr[bfi] + 32, gb + 32);
        async_b128(ldsBaddr[bfi] + 48, gb + 48);
        __builtin_prefetch(bSrc + k0 + 2 * KSTEP, 0, 1);
    };

    int aOff[2], bOff[4];
    for (int mi = 0; mi < 2; mi++) aOff[mi] = (wr * 32 + mi * 16 + r0) * LROW;
    for (int ni = 0; ni < 4; ni++) bOff[ni] = (wc * 64 + ni * 16 + r0) * LROW;

    v8f acc[2][4];
    for (int mi = 0; mi < 2; mi++)
        for (int ni = 0; ni < 4; ni++)
            acc[mi][ni] = (v8f){0.f, 0.f, 0.f, 0.f, 0.f, 0.f, 0.f, 0.f};

    const int NK = DFF / KSTEP;
    stage(0, 0);
    for (int kk = 0; kk < NK; kk++) {
        int cur = kk & 1;
        if (kk + 1 < NK) {
            stage(cur ^ 1, (kk + 1) * KSTEP);
            asm volatile("s_wait_asynccnt 0x8" ::: "memory");
        } else {
            asm volatile("s_wait_asynccnt 0x0" ::: "memory");
        }
        __syncthreads();
        for (int ks = 0; ks < 2; ks++) {
            int kso = ks * 32;
            FragBF a[2], bfr[4];
            for (int mi = 0; mi < 2; mi++) {
                a[mi].q[0] = *(const uint4*)&ldsA[cur][aOff[mi] + kso + half * 8];
                a[mi].q[1] = *(const uint4*)&ldsA[cur][aOff[mi] + kso + 16 + half * 8];
            }
            for (int ni = 0; ni < 4; ni++) {
                bfr[ni].q[0] = *(const uint4*)&ldsB[cur][bOff[ni] + kso + half * 8];
                bfr[ni].q[1] = *(const uint4*)&ldsB[cur][bOff[ni] + kso + 16 + half * 8];
            }
            for (int ni = 0; ni < 4; ni++)
                for (int mi = 0; mi < 2; mi++)
                    acc[mi][ni] = __builtin_amdgcn_wmma_f32_16x16x32_bf16(
                        false, a[mi].bf, false, bfr[ni].bf, (short)0, acc[mi][ni], false, false);
        }
        __syncthreads();
    }

    for (int mi = 0; mi < 2; mi++)
        for (int ni = 0; ni < 4; ni++) {
            int n = nBase + wc * 64 + ni * 16 + r0;
            float bias = eb2[e * D + n];
            for (int i = 0; i < 8; i++) {
                int l = wr * 32 + mi * 16 + i + 8 * half;
                int slot = mBase + l;
                if (slot < cnt) {
                    int tok = sTok[l];
                    float g = combine[(size_t)tok * EXPERTS + e];
                    size_t oi = (size_t)tok * D + n;
                    out[oi] = out[oi] + g * (acc[mi][ni][i] + bias);  // rows unique per expert; experts serialized
                }
            }
        }
}

// ---------------------------------------------------------------- lb finalize
__global__ void finalize_lb_kernel(const float* f_acc, const float* p_acc, float* out) {
    if (threadIdx.x == 0) {
        float s = 0.f;
        for (int e = 0; e < EXPERTS; e++)
            s += (f_acc[e] / (float)T) * (p_acc[e] / (float)T);
        out[(size_t)T * D] = (float)EXPERTS * s;
    }
}

// ================================================================= launch
extern "C" void kernel_launch(void* const* d_in, const int* in_sizes, int n_in,
                              void* d_out, int out_size, void* d_ws, size_t ws_size,
                              hipStream_t stream) {
    (void)in_sizes; (void)n_in; (void)out_size; (void)ws_size;
    const float* tokens  = (const float*)d_in[0];
    const float* alt_emb = (const float*)d_in[1];
    const float* gamma   = (const float*)d_in[2];
    const float* beta    = (const float*)d_in[3];
    const float* gw1     = (const float*)d_in[4];
    const float* gb1     = (const float*)d_in[5];
    const float* gw2     = (const float*)d_in[6];
    const float* gb2     = (const float*)d_in[7];
    const float* ew1     = (const float*)d_in[8];
    const float* eb1     = (const float*)d_in[9];
    const float* ew2     = (const float*)d_in[10];
    const float* eb2     = (const float*)d_in[11];
    float* out = (float*)d_out;

    char* p = (char*)d_ws;
    auto alloc = [&](size_t bytes) -> void* {
        void* r = (void*)p;
        p += (bytes + 255) & ~(size_t)255;
        return r;
    };
    unsigned short* normed = (unsigned short*)alloc((size_t)T * D * 2);
    unsigned short* w1t    = (unsigned short*)alloc((size_t)EXPERTS * DFF * D * 2);
    unsigned short* w2t    = (unsigned short*)alloc((size_t)EXPERTS * D * DFF * 2);
    unsigned short* hbuf   = (unsigned short*)alloc((size_t)T * DFF * 2);
    float* combine = (float*)alloc((size_t)T * EXPERTS * 4);
    int*   lists   = (int*)alloc((size_t)EXPERTS * T * 4);
    int*   counts  = (int*)alloc(64);
    float* f_acc   = (float*)alloc(64);
    float* p_acc   = (float*)alloc(64);

    init_small_kernel<<<1, 64, 0, stream>>>(counts, f_acc, p_acc);
    copy_tokens_kernel<<<2048, 256, 0, stream>>>(tokens, out, T * D);

    dim3 tb(32, 8);
    transpose_f32_to_bf16<<<dim3(DFF / 32, D / 32, EXPERTS), tb, 0, stream>>>(ew1, w1t, D, DFF);
    transpose_f32_to_bf16<<<dim3(D / 32, DFF / 32, EXPERTS), tb, 0, stream>>>(ew2, w2t, DFF, D);

    ln_router_kernel<<<T, 256, 0, stream>>>(tokens, alt_emb, gamma, beta,
                                            gw1, gb1, gw2, gb2,
                                            normed, combine, counts, lists, f_acc, p_acc);

    for (int e = 0; e < EXPERTS; e++) {
        expert_gemm1_kernel<<<dim3(DFF / 128, T / 128), 256, 0, stream>>>(
            normed, w1t, eb1, counts, lists, hbuf, e);
        expert_gemm2_kernel<<<dim3(D / 128, T / 128), 256, 0, stream>>>(
            hbuf, w2t, eb2, combine, counts, lists, out, e);
    }

    finalize_lb_kernel<<<1, 32, 0, stream>>>(f_acc, p_acc, out);
}
